// VerifyLlamaAttention_3848290697308
// MI455X (gfx1250) — matrix operand order repaired and checked by reading, not compiled, not run on hardware
//
#include <hip/hip_runtime.h>
#include <hip/hip_bf16.h>

// Problem constants (match reference)
#define B_    2
#define S_    2048
#define D_    4096
#define H_    32
#define KVH_  8
#define HD_   128
#define NREP_ 4
#define SCALE_ 0.08838834764831845f   // 128^-0.5

typedef __attribute__((ext_vector_type(16))) _Float16 v16h;
typedef __attribute__((ext_vector_type(8)))  _Float16 v8h;
typedef __attribute__((ext_vector_type(4)))  _Float16 v4h;
typedef __attribute__((ext_vector_type(8)))  float    v8f;
typedef __attribute__((ext_vector_type(4)))  int      v4i;

// Address-space-qualified int4 pointers for the async global->LDS builtin:
// clang prints AS1 as "__device__" and AS3 as "__shared__" in HIP diagnostics.
typedef __attribute__((address_space(1))) v4i as1_v4i;
typedef __attribute__((address_space(3))) v4i as3_v4i;

// CDNA5 async global->LDS copy (ASYNCcnt path), with sync fallback.
#if defined(__has_builtin)
#  if __has_builtin(__builtin_amdgcn_global_load_async_to_lds_b128)
#    define ASYNC_LDS 1
#  else
#    define ASYNC_LDS 0
#  endif
#  if __has_builtin(__builtin_amdgcn_s_wait_asynccnt)
#    define WAIT_ASYNC_BUILTIN 1
#  else
#    define WAIT_ASYNC_BUILTIN 0
#  endif
#else
#  define ASYNC_LDS 0
#  define WAIT_ASYNC_BUILTIN 0
#endif

__device__ __forceinline__ void wait_async0() {
#if WAIT_ASYNC_BUILTIN
  __builtin_amdgcn_s_wait_asynccnt(0);
#else
  asm volatile("s_wait_asynccnt 0x0" ::: "memory");
#endif
}

union Frag { v16h v; v8h h[2]; };

enum { MODE_Q = 0, MODE_K, MODE_V, MODE_SCORES, MODE_PV, MODE_OUT };

static constexpr int BM  = 128;   // block tile M
static constexpr int BN  = 128;   // block tile N
static constexpr int BK  = 32;    // K per WMMA step (== K of v_wmma_f32_16x16x32_f16)
static constexpr int LDT = 40;    // padded LDS row stride (halves): 80B rows keep 16B align

// Stage a 128x32 tile from row-major global (leading dim = ld) into LDS as f16.
// float path: batch 4 global_load_b128 into regs, then cvt + ds_store (MLP-friendly).
// f16 path: async global->LDS b128 copies when available (no VGPR round-trip).
template<typename T>
__device__ __forceinline__ void load_tile(const T* __restrict__ g, int ld,
                                          _Float16* lds, int tid) {
  if constexpr (sizeof(T) == 4) {
    float4 f[4];
#pragma unroll
    for (int i = 0; i < 4; ++i) {
      int idx = tid + i * 256;              // 1024 float4 chunks, 8 per row
      f[i] = *(const float4*)(g + (size_t)(idx >> 3) * ld + ((idx & 7) << 2));
    }
#pragma unroll
    for (int i = 0; i < 4; ++i) {
      int idx = tid + i * 256;
      int r = idx >> 3, c = (idx & 7) << 2;
      v4h h = { (_Float16)f[i].x, (_Float16)f[i].y, (_Float16)f[i].z, (_Float16)f[i].w };
      *(v4h*)(lds + r * LDT + c) = h;       // 8B ds_store
    }
  } else {
#if ASYNC_LDS
#pragma unroll
    for (int i = 0; i < 2; ++i) {
      int idx = tid + i * 256;              // 512 8-half chunks, 4 per row
      int r = idx >> 2, c = (idx & 3) << 3;
      __builtin_amdgcn_global_load_async_to_lds_b128(
          (as1_v4i*)(g + (size_t)r * ld + c),
          (as3_v4i*)(lds + r * LDT + c), 0, 0);
    }
#else
    v8h t[2];
#pragma unroll
    for (int i = 0; i < 2; ++i) {
      int idx = tid + i * 256;
      t[i] = *(const v8h*)(g + (size_t)(idx >> 2) * ld + ((idx & 3) << 3));
    }
#pragma unroll
    for (int i = 0; i < 2; ++i) {
      int idx = tid + i * 256;
      *(v8h*)(lds + (idx >> 2) * LDT + ((idx & 3) << 3)) = t[i];
    }
#endif
  }
}

// One GEMM for everything: C(MxN) = A(MxK, row-major) * B(NxK, row-major)^T [+ bias]
// 256 threads = 8 wave32 waves (2x4 grid); each wave owns 64x32 (4x2 WMMA tiles).
// Double-buffered LDS, one barrier per K-step, epilogue differs per MODE.
template<int MODE, typename TA, typename TB>
__global__ __launch_bounds__(256) void gemm_wmma(
    const TA* __restrict__ Abase, const TB* __restrict__ Bbase,
    const float* __restrict__ bias, void* __restrict__ Cbase,
    const float* __restrict__ mask, int M, int N, int K)
{
  (void)M; (void)bias; (void)mask;
  __shared__ __align__(16) _Float16 sA[2][BM * LDT];
  __shared__ __align__(16) _Float16 sB[2][BN * LDT];

  const int tid  = threadIdx.x;
  const int lane = tid & 31;
  const int wave = tid >> 5;
  const int wm   = wave >> 2;          // 0..1  (64 rows each)
  const int wn   = wave & 3;           // 0..3  (32 cols each)
  const int lrow = lane & 15;
  const int hi   = lane >> 4;          // half-wave select per ISA VGPR layout
  const int row0 = blockIdx.y * BM;
  const int col0 = blockIdx.x * BN;
  const int z    = blockIdx.z;

  int zb = 0, zh = 0;
  const TA* A  = Abase;
  const TB* Bm = Bbase;
  if constexpr (MODE == MODE_SCORES) {
    zb = z / H_; zh = z % H_;
    A  = Abase + (size_t)z * S_ * HD_;                          // Q head [S][HD]
    Bm = Bbase + (size_t)(zb * KVH_ + zh / NREP_) * S_ * HD_;   // K head [S][HD] (GQA)
  } else if constexpr (MODE == MODE_PV) {
    zb = z / H_; zh = z % H_;
    A  = Abase + (size_t)z * S_ * S_;                           // P head [S][S] fp32
    Bm = Bbase + (size_t)(zb * KVH_ + zh / NREP_) * HD_ * S_;   // V^T head [HD][S]
  }

  constexpr bool useAsync = ASYNC_LDS && (sizeof(TA) == 2 || sizeof(TB) == 2);

  v8f acc[4][2] = {};

  // Prologue: fill buffer 0
  load_tile<TA>(A  + (size_t)row0 * K, K, sA[0], tid);
  load_tile<TB>(Bm + (size_t)col0 * K, K, sB[0], tid);

  int cur = 0;
  for (int k0 = 0; k0 < K; k0 += BK) {
    if constexpr (useAsync) wait_async0();   // drain this wave's async LDS fills
    __syncthreads();                         // all waves' fills visible

    const int nxt = cur ^ 1;
    if (k0 + BK < K) {                       // fill next buffer under current compute
      load_tile<TA>(A  + (size_t)row0 * K + k0 + BK, K, sA[nxt], tid);
      load_tile<TB>(Bm + (size_t)col0 * K + k0 + BK, K, sB[nxt], tid);
    }
    if (k0 + 2 * BK < K) {                   // global_prefetch_b8 two tiles ahead
      __builtin_prefetch(A  + (size_t)(row0 + (tid & (BM - 1))) * K + k0 + 2 * BK, 0, 1);
      __builtin_prefetch(Bm + (size_t)(col0 + (tid & (BN - 1))) * K + k0 + 2 * BK, 0, 1);
    }

    // Fragment loads matching ISA 16-bit A(16x32) / B(32x16) wave32 layouts.
    const _Float16* bufA = sA[cur];
    const _Float16* bufB = sB[cur];
    Frag fa[4], fb[2];
#pragma unroll
    for (int mi = 0; mi < 4; ++mi) {
      const _Float16* pa = &bufA[(wm * 64 + mi * 16 + lrow) * LDT + hi * 8];
      fa[mi].h[0] = *(const v8h*)pa;          // K 0..7  (hi=0)  / 8..15  (hi=1)
      fa[mi].h[1] = *(const v8h*)(pa + 16);   // K 16..23 (hi=0) / 24..31 (hi=1)
    }
#pragma unroll
    for (int ni = 0; ni < 2; ++ni) {
      const _Float16* pb = &bufB[(wn * 32 + ni * 16 + lrow) * LDT + hi * 8];
      fb[ni].h[0] = *(const v8h*)pb;          // K 0..7  (hi=0)  / 8..15  (hi=1)
      fb[ni].h[1] = *(const v8h*)(pb + 16);   // K 16..23 (hi=0) / 24..31 (hi=1)
    }
#pragma unroll
    for (int mi = 0; mi < 4; ++mi)
#pragma unroll
      for (int ni = 0; ni < 2; ++ni)
        acc[mi][ni] = __builtin_amdgcn_wmma_f32_16x16x32_f16(
            false, fa[mi].v, false, fb[ni].v, (short)0, acc[mi][ni], false, false);
    cur = nxt;
  }

  // Epilogue: C/D layout — lane holds col = lrow, rows = hi*8 + j (j = vreg index).
#pragma unroll
  for (int mi = 0; mi < 4; ++mi) {
#pragma unroll
    for (int ni = 0; ni < 2; ++ni) {
#pragma unroll
      for (int j = 0; j < 8; ++j) {
        int row = row0 + wm * 64 + mi * 16 + hi * 8 + j;
        int col = col0 + wn * 32 + ni * 16 + lrow;
        float v = acc[mi][ni][j];
        if constexpr (MODE == MODE_OUT) {
          v += bias[col];
          __builtin_nontemporal_store(v, (float*)Cbase + (size_t)row * N + col);
        } else if constexpr (MODE == MODE_Q) {
          v += bias[col];
          int b = row >> 11, s = row & (S_ - 1);
          int h = col >> 7,  hd = col & (HD_ - 1);
          ((_Float16*)Cbase)[(((size_t)(b * H_ + h)) * S_ + s) * HD_ + hd] = (_Float16)v;
        } else if constexpr (MODE == MODE_K) {
          v += bias[col];
          int b = row >> 11, s = row & (S_ - 1);
          int h = col >> 7,  hd = col & (HD_ - 1);
          ((_Float16*)Cbase)[(((size_t)(b * KVH_ + h)) * S_ + s) * HD_ + hd] = (_Float16)v;
        } else if constexpr (MODE == MODE_V) {
          v += bias[col];
          int b = row >> 11, s = row & (S_ - 1);
          int h = col >> 7,  hd = col & (HD_ - 1);
          // store V transposed: [B][KVH][HD][S] so P·V is a plain X·W^T GEMM
          ((_Float16*)Cbase)[(((size_t)(b * KVH_ + h)) * HD_ + hd) * S_ + s] = (_Float16)v;
        } else if constexpr (MODE == MODE_SCORES) {
          float m = mask[((size_t)zb * S_ + row) * S_ + col];
          float* pc = (float*)Cbase + (size_t)z * S_ * S_ + (size_t)row * S_ + col;
          __builtin_nontemporal_store((v + m) * SCALE_, pc);  // streaming 1GB tensor
        } else { // MODE_PV: ctx token-major f16 [B][S][H*HD]
          ((_Float16*)Cbase)[(((size_t)(zb * S_ + row)) * H_ + zh) * HD_ + col] = (_Float16)v;
        }
      }
    }
  }
}

// RoPE in-place on f16 head-major tensor [B][NH][S][HD]; one thread per pair.
__global__ __launch_bounds__(256) void rope_kernel(
    _Float16* __restrict__ x, const float* __restrict__ cosp,
    const float* __restrict__ sinp, int NH, size_t total)
{
  size_t i = (size_t)blockIdx.x * blockDim.x + threadIdx.x;
  if (i >= total) return;
  int hd = (int)(i & 63);
  size_t t = i >> 6;
  int s = (int)(t % S_); t /= S_;
  int h = (int)(t % NH);
  int b = (int)(t / NH);
  size_t base = (((size_t)(b * NH + h)) * S_ + s) * HD_;
  const float* cp = cosp + ((size_t)b * S_ + s) * HD_;
  const float* sp = sinp + ((size_t)b * S_ + s) * HD_;
  float x1 = (float)x[base + hd];
  float x2 = (float)x[base + hd + 64];
  float o1 = x1 * cp[hd]      - x2 * sp[hd];        // x*cos + rot_half(x)*sin
  float o2 = x2 * cp[hd + 64] + x1 * sp[hd + 64];
  x[base + hd]      = (_Float16)o1;
  x[base + hd + 64] = (_Float16)o2;
}

// Softmax over rows of length S_ (2048), in place, fp32. One wave per row,
// 64 values per lane in registers; shfl_xor reductions (wave32).
__global__ __launch_bounds__(256) void softmax_rows(float* __restrict__ p)
{
  size_t row = (size_t)blockIdx.x * 8 + (threadIdx.x >> 5);
  int lane = threadIdx.x & 31;
  float* r = p + row * S_;
  float v[64];
  float mx = -3.4e38f;
#pragma unroll
  for (int i = 0; i < 16; ++i) {
    float4 f = ((const float4*)r)[lane + i * 32];
    v[i * 4 + 0] = f.x; v[i * 4 + 1] = f.y; v[i * 4 + 2] = f.z; v[i * 4 + 3] = f.w;
    mx = fmaxf(mx, fmaxf(fmaxf(f.x, f.y), fmaxf(f.z, f.w)));
  }
#pragma unroll
  for (int off = 16; off > 0; off >>= 1)
    mx = fmaxf(mx, __shfl_xor(mx, off, 32));
  float sum = 0.f;
#pragma unroll
  for (int j = 0; j < 64; ++j) { v[j] = __expf(v[j] - mx); sum += v[j]; }
#pragma unroll
  for (int off = 16; off > 0; off >>= 1)
    sum += __shfl_xor(sum, off, 32);
  float inv = 1.0f / sum;
#pragma unroll
  for (int i = 0; i < 16; ++i) {
    float4 f = { v[i * 4] * inv, v[i * 4 + 1] * inv, v[i * 4 + 2] * inv, v[i * 4 + 3] * inv };
    ((float4*)r)[lane + i * 32] = f;
  }
}

extern "C" void kernel_launch(void* const* d_in, const int* in_sizes, int n_in,
                              void* d_out, int out_size, void* d_ws, size_t ws_size,
                              hipStream_t stream)
{
  (void)in_sizes; (void)n_in; (void)out_size; (void)ws_size;
  const float* hidden = (const float*)d_in[0];
  const float* cosp   = (const float*)d_in[1];
  const float* sinp   = (const float*)d_in[2];
  const float* mask   = (const float*)d_in[3];
  const float* wq = (const float*)d_in[4];
  const float* bq = (const float*)d_in[5];
  const float* wk = (const float*)d_in[6];
  const float* bk = (const float*)d_in[7];
  const float* wv = (const float*)d_in[8];
  const float* bv = (const float*)d_in[9];
  const float* wo = (const float*)d_in[10];
  const float* bo = (const float*)d_in[11];

  float* out = (float*)d_out;                       // [B,S,D] fp32
  float* p   = out + (size_t)B_ * S_ * D_;          // [B,H,S,S] fp32 (output #2)

  // Workspace: ~84MB of f16 intermediates
  _Float16* q16 = (_Float16*)d_ws;                              // [B][H][S][HD]
  _Float16* k16 = q16 + (size_t)B_ * H_   * S_  * HD_;          // [B][KVH][S][HD]
  _Float16* v16 = k16 + (size_t)B_ * KVH_ * S_  * HD_;          // [B][KVH][HD][S]
  _Float16* c16 = v16 + (size_t)B_ * KVH_ * HD_ * S_;           // [B][S][H*HD]

  dim3 blk(256);
  const int Mtok = B_ * S_;  // 4096

  // QKV projections (fp32 in -> f16 out, scattered to attention-friendly layouts)
  gemm_wmma<MODE_Q, float, float><<<dim3(D_ / BN, Mtok / BM, 1), blk, 0, stream>>>(
      hidden, wq, bq, q16, nullptr, Mtok, H_ * HD_, D_);
  gemm_wmma<MODE_K, float, float><<<dim3((KVH_ * HD_) / BN, Mtok / BM, 1), blk, 0, stream>>>(
      hidden, wk, bk, k16, nullptr, Mtok, KVH_ * HD_, D_);
  gemm_wmma<MODE_V, float, float><<<dim3((KVH_ * HD_) / BN, Mtok / BM, 1), blk, 0, stream>>>(
      hidden, wv, bv, v16, nullptr, Mtok, KVH_ * HD_, D_);

  // RoPE in place on q and k
  {
    size_t nq = (size_t)B_ * H_   * S_ * (HD_ / 2);
    size_t nk = (size_t)B_ * KVH_ * S_ * (HD_ / 2);
    rope_kernel<<<dim3((unsigned)((nq + 255) / 256)), blk, 0, stream>>>(q16, cosp, sinp, H_,   nq);
    rope_kernel<<<dim3((unsigned)((nk + 255) / 256)), blk, 0, stream>>>(k16, cosp, sinp, KVH_, nk);
  }

  // scores = (Q K^T + mask) * scale  -> p region (fp32, streaming NT stores)
  gemm_wmma<MODE_SCORES, _Float16, _Float16><<<dim3(S_ / BN, S_ / BM, B_ * H_), blk, 0, stream>>>(
      q16, k16, nullptr, p, mask, S_, S_, HD_);

  // softmax in place over last axis
  softmax_rows<<<dim3((unsigned)((size_t)B_ * H_ * S_ / 8)), blk, 0, stream>>>(p);

  // ctx = P V  (P fp32 read+cvt, V^T f16) -> ctx token-major f16
  gemm_wmma<MODE_PV, float, _Float16><<<dim3(HD_ / BN, S_ / BM, B_ * H_), blk, 0, stream>>>(
      p, v16, nullptr, c16, nullptr, S_, HD_, S_);

  // out = ctx Wo^T + bo -> d_out fp32
  gemm_wmma<MODE_OUT, _Float16, float><<<dim3(D_ / BN, Mtok / BM, 1), blk, 0, stream>>>(
      c16, wo, bo, out, nullptr, Mtok, D_, D_);
}
